// KullbackLeiblerDivergence_18305150615592
// MI455X (gfx1250) — matrix operand order repaired and checked by reading, not compiled, not run on hardware
//
#include <hip/hip_runtime.h>
#include <math.h>

// ---------------------------------------------------------------------------
// Pairwise KL divergence: out[n,m] = ent[n] - sum_d a[n,d] * log(b[m,d])
// N = M = 8192, D = 64.
//
// MI455X reasoning: output stores (256 MB) dominate -> ~11us at 23.3 TB/s.
// GEMM is 8.6 GFLOP; to stay store-bound we need ~780 TFLOPS effective, so we
// use bf16 WMMA (16x16x32) with a 2-term bf16 split (hi+lo) of both operands
// for near-f32 accuracy at 3x bf16-WMMA cost (still well under the HBM bound).
// ---------------------------------------------------------------------------

typedef __attribute__((ext_vector_type(16))) __bf16 v16bf;
typedef __attribute__((ext_vector_type(8)))  __bf16 v8bf;
typedef __attribute__((ext_vector_type(8)))  float  v8f;

static constexpr int NDIM = 8192;
static constexpr int MDIM = 8192;
static constexpr int DDIM = 64;

// ------------------------------- prep kernels ------------------------------

__global__ void kl_ent_kernel(const float* __restrict__ a,
                              float* __restrict__ ent) {
  int n = blockIdx.x * blockDim.x + threadIdx.x;
  if (n >= NDIM) return;
  const float* row = a + (size_t)n * DDIM;
  float s = 0.f;
#pragma unroll 8
  for (int d = 0; d < DDIM; ++d) {
    float v = row[d];
    s += v * logf(v);
  }
  ent[n] = s;
}

// split a[i] = hi + lo (bf16 pair, ~16 mantissa bits total)
__global__ void kl_split_a_kernel(const float* __restrict__ a,
                                  __bf16* __restrict__ hi,
                                  __bf16* __restrict__ lo) {
  int i = blockIdx.x * blockDim.x + threadIdx.x;
  if (i >= NDIM * DDIM) return;
  float v = a[i];
  __bf16 h = (__bf16)v;
  float r = v - (float)h;   // exact in f32
  hi[i] = h;
  lo[i] = (__bf16)r;
}

// split (-log b[i]) = hi + lo; negation lets the GEMM accumulate with plain
// adds on top of an accumulator preloaded with ent[n] (bf16 WMMA has no A/B
// negate modifier per the ISA NEG table).
__global__ void kl_split_nlogb_kernel(const float* __restrict__ b,
                                      __bf16* __restrict__ hi,
                                      __bf16* __restrict__ lo) {
  int i = blockIdx.x * blockDim.x + threadIdx.x;
  if (i >= MDIM * DDIM) return;
  float v = -logf(b[i]);
  __bf16 h = (__bf16)v;
  float r = v - (float)h;
  hi[i] = h;
  lo[i] = (__bf16)r;
}

// ------------------------------- GEMM kernel -------------------------------

__device__ __forceinline__ v16bf load_frag16(const __bf16* p0, const __bf16* p1) {
  v8bf x = *(const v8bf*)p0;   // global_load_b128
  v8bf y = *(const v8bf*)p1;   // global_load_b128
  return __builtin_shufflevector(x, y, 0, 1, 2, 3, 4, 5, 6, 7,
                                        8, 9, 10, 11, 12, 13, 14, 15);
}

__device__ __forceinline__ v8f wmma_bf16(v16bf a, v16bf b, v8f c) {
  return __builtin_amdgcn_wmma_f32_16x16x32_bf16(
      /*neg_a=*/false, a, /*neg_b=*/false, b,
      /*c_mod=*/(short)0, c, /*reuse_a=*/false, /*reuse_b=*/false);
}

// One wave computes a 16(n) x 64(m) output strip: A fragments loaded once and
// reused across 4 consecutive 16x16 m-tiles. 8 waves (256 thr) per block.
__global__ void __launch_bounds__(256)
kl_wmma_kernel(const __bf16* __restrict__ Ah, const __bf16* __restrict__ Al,
               const __bf16* __restrict__ Bh, const __bf16* __restrict__ Bl,
               const float* __restrict__ ent, float* __restrict__ out) {
  const int lane  = threadIdx.x & 31;
  const int wave  = threadIdx.x >> 5;
  const int strip = blockIdx.x * 8 + wave;       // 512 n-tiles * 128 m-strips
  const int n_tile  = strip >> 7;
  const int m_strip = strip & 127;
  const int n0  = n_tile * 16;
  const int g   = lane >> 4;                     // half-wave group
  const int l15 = lane & 15;

  // A fragment VGPR layout (16-bit A 16x32): lane group g: V0-3 <- K = 8g..8g+7,
  // V4-7 <- K = 16+8g..16+8g+7; row M = lane&15. Two K-chunks cover D=64.
  const __bf16* a_h = Ah + (size_t)(n0 + l15) * DDIM;
  const __bf16* a_l = Al + (size_t)(n0 + l15) * DDIM;
  v16bf ah0 = load_frag16(a_h +      8 * g, a_h + 16 + 8 * g);
  v16bf ah1 = load_frag16(a_h + 32 + 8 * g, a_h + 48 + 8 * g);
  v16bf al0 = load_frag16(a_l +      8 * g, a_l + 16 + 8 * g);
  v16bf al1 = load_frag16(a_l + 32 + 8 * g, a_l + 48 + 8 * g);

  // C/D layout: lane -> N = lane&15, VGPR v -> M = v + 8g. Preload ent rows.
  float e[8];
#pragma unroll
  for (int v = 0; v < 8; ++v) e[v] = ent[n0 + 8 * g + v];

#pragma unroll
  for (int t = 0; t < 4; ++t) {
    const int m0 = (m_strip * 4 + t) * 16;
    // B fragment (16-bit B 32x16): col N=lane&15; lane group g: elems 0-7 <-
    // K = 8g..8g+7, elems 8-15 <- K = 16+8g..16+8g+7 (same order as A).
    const __bf16* b_h = Bh + (size_t)(m0 + l15) * DDIM;
    const __bf16* b_l = Bl + (size_t)(m0 + l15) * DDIM;
    v16bf bh0 = load_frag16(b_h +      8 * g, b_h + 16 + 8 * g);
    v16bf bh1 = load_frag16(b_h + 32 + 8 * g, b_h + 48 + 8 * g);
    v16bf bl0 = load_frag16(b_l +      8 * g, b_l + 16 + 8 * g);
    v16bf bl1 = load_frag16(b_l + 32 + 8 * g, b_l + 48 + 8 * g);

    v8f acc;
#pragma unroll
    for (int v = 0; v < 8; ++v) acc[v] = e[v];

    // out = ent + a*(-log b): split-product terms hi*hi + hi*lo + lo*hi.
    acc = wmma_bf16(ah0, bh0, acc);
    acc = wmma_bf16(ah1, bh1, acc);
    acc = wmma_bf16(ah0, bl0, acc);
    acc = wmma_bf16(ah1, bl1, acc);
    acc = wmma_bf16(al0, bh0, acc);
    acc = wmma_bf16(al1, bh1, acc);

    float* orow = out + (size_t)(n0 + 8 * g) * MDIM + m0 + l15;
#pragma unroll
    for (int v = 0; v < 8; ++v) orow[(size_t)v * MDIM] = acc[v];
  }
}

// ------------------------------- launcher ----------------------------------

extern "C" void kernel_launch(void* const* d_in, const int* in_sizes, int n_in,
                              void* d_out, int out_size, void* d_ws, size_t ws_size,
                              hipStream_t stream) {
  const float* a = (const float*)d_in[0];
  const float* b = (const float*)d_in[1];
  float* out = (float*)d_out;

  // workspace layout: ent (32KB) | Ah | Al | Bh | Bl  (1MB each, ~4.03MB total)
  char* ws = (char*)d_ws;
  float*  ent = (float*)ws;
  __bf16* Ah  = (__bf16*)(ws + 32768);
  __bf16* Al  = (__bf16*)(ws + 32768 + 1 * 1048576);
  __bf16* Bh  = (__bf16*)(ws + 32768 + 2 * 1048576);
  __bf16* Bl  = (__bf16*)(ws + 32768 + 3 * 1048576);

  kl_ent_kernel<<<NDIM / 256, 256, 0, stream>>>(a, ent);
  kl_split_a_kernel<<<(NDIM * DDIM) / 256, 256, 0, stream>>>(a, Ah, Al);
  kl_split_nlogb_kernel<<<(MDIM * DDIM) / 256, 256, 0, stream>>>(b, Bh, Bl);

  // 512 n-tiles * 128 m-strips = 65536 strips; 8 waves/block -> 8192 blocks
  kl_wmma_kernel<<<(512 * 128) / 8, 256, 0, stream>>>(Ah, Al, Bh, Bl, ent, out);
}
